// SoftModularizationNetwork_47382079209770
// MI455X (gfx1250) — matrix operand order repaired and checked by reading, not compiled, not run on hardware
//
#include <hip/hip_runtime.h>

// ---------------- constants from the reference ----------------
#define BB    8192
#define OBS   128
#define NTASK 50
#define EE    512
#define DD    512
#define LL    4
#define MM    8
#define RE    256

#define LDS_STRIDE 40   // padded bf16 row stride: banks (row*20)%64 all distinct

typedef __attribute__((ext_vector_type(16))) __bf16 v16bf;
typedef __attribute__((ext_vector_type(8)))  float  v8f;

union FragBF { v16bf v; uint4 q[2]; };

// ---- async Global -> LDS 16B copy (ASYNCcnt-tracked) -----------------------
__device__ __forceinline__ void async_cp16(unsigned lds_off, const void* gptr) {
  asm volatile("global_load_async_to_lds_b128 %0, %1, off"
               :: "v"(lds_off), "v"(gptr)
               : "memory");
}

__device__ __forceinline__ void wait_async0() {
#if defined(__has_builtin) && __has_builtin(__builtin_amdgcn_s_wait_asynccnt)
  __builtin_amdgcn_s_wait_asynccnt(0);
#else
  asm volatile("s_wait_asynccnt 0x0" ::: "memory");
#endif
}

// ---- A fragment from LDS (bf16 16x32) --------------------------------------
// ISA 7.12.2: lanes 0-15 -> K = {0..7, 16..23}, lanes 16-31 -> K = {8..15, 24..31}
__device__ __forceinline__ v16bf lds_a_frag(const __bf16* buf, int row0, int lane) {
  int m    = row0 + (lane & 15);
  int half = (lane >> 4) & 1;
  const __bf16* p = buf + m * LDS_STRIDE + half * 8;
  FragBF f;
  f.q[0] = *(const uint4*)(p);        // K = half*8    .. +7
  f.q[1] = *(const uint4*)(p + 16);   // K = half*8+16 .. +23
  return f.v;
}

// ---- B fragment from LDS (bf16 32x16, tile stored [col][K]) ----------------
// lanes 0-15 hold K = {0..7, 16..23} of column n, lanes 16-31 hold K = {8..15, 24..31}
__device__ __forceinline__ v16bf lds_b_frag(const __bf16* buf, int n0, int lane) {
  int n    = n0 + (lane & 15);
  int half = (lane >> 4) & 1;
  const __bf16* p = buf + n * LDS_STRIDE + half * 8;
  FragBF f;
  f.q[0] = *(const uint4*)(p);
  f.q[1] = *(const uint4*)(p + 16);
  return f.v;
}

// issue one 128x32 A tile + 128x32 B tile into LDS (4 async b128 per thread)
__device__ __forceinline__ void issue_tile(const __bf16* __restrict__ Ag, int lda,
                                           const __bf16* __restrict__ Bg, int K, int k0,
                                           unsigned ldsA, unsigned ldsB, int t) {
#pragma unroll
  for (int r = 0; r < 2; ++r) {
    int id  = t + r * 256;
    int row = id >> 2;
    int ch  = (id & 3) * 8;     // 8 bf16 = 16 bytes per chunk
    async_cp16(ldsA + (unsigned)(row * LDS_STRIDE + ch) * 2,
               Ag + (long)row * lda + k0 + ch);
    async_cp16(ldsB + (unsigned)(row * LDS_STRIDE + ch) * 2,
               Bg + (long)row * K + k0 + ch);
  }
}

// ---------------- generic bf16 WMMA GEMM (LDS double-buffered) --------------
// C[rows x N] = A[rows x K] @ Wt^T (+ bias), Wt stored [N][K] bf16.
// Block: 256 thr = 8 waves, waves arranged 4(M) x 2(N): tile 128 x 128.
// blockIdx.z = module index with per-module pointer strides (0 if unused).
__global__ void __launch_bounds__(256)
gemm_bf16_wmma(const __bf16* __restrict__ A, int lda, long a_mod,
               const __bf16* __restrict__ Wt, int K, long w_mod,
               const float* __restrict__ bias, long b_mod,
               float* __restrict__ Cf, int ldcf, long cf_mod, int relu_f,
               __bf16* __restrict__ Cb, int ldcb, long cb_mod, int relu_b) {
  __shared__ __align__(16) __bf16 sA[2][128 * LDS_STRIDE];
  __shared__ __align__(16) __bf16 sB[2][128 * LDS_STRIDE];

  int t    = threadIdx.x;
  int lane = t & 31;
  int wave = t >> 5;
  int wm = wave & 3;            // row sub-tile (32 rows each)
  int wn = wave >> 2;           // col sub-tile (64 cols each)
  int z  = blockIdx.z;

  const __bf16* Ag = A  + (long)z * a_mod + (long)blockIdx.x * 128 * lda;
  const __bf16* Bg = Wt + (long)z * w_mod + (long)blockIdx.y * 128 * K;
  if (bias) bias += (long)z * b_mod;

  unsigned baseA0 = (unsigned)(size_t)&sA[0][0];
  unsigned baseA1 = (unsigned)(size_t)&sA[1][0];
  unsigned baseB0 = (unsigned)(size_t)&sB[0][0];
  unsigned baseB1 = (unsigned)(size_t)&sB[1][0];

  issue_tile(Ag, lda, Bg, K, 0, baseA0, baseB0, t);

  v8f acc[2][4] = {};
  int T = K >> 5;
  for (int kt = 0; kt < T; ++kt) {
    wait_async0();              // own tile-kt copies done
    __syncthreads();            // everyone's copies visible
    if (kt + 1 < T)             // overlap next copy with this tile's math
      issue_tile(Ag, lda, Bg, K, (kt + 1) * 32,
                 ((kt + 1) & 1) ? baseA1 : baseA0,
                 ((kt + 1) & 1) ? baseB1 : baseB0, t);

    const __bf16* as = sA[kt & 1];
    const __bf16* bs = sB[kt & 1];
    v16bf a0 = lds_a_frag(as, wm * 32,      lane);
    v16bf a1 = lds_a_frag(as, wm * 32 + 16, lane);
    v16bf b0 = lds_b_frag(bs, wn * 64,      lane);
    v16bf b1 = lds_b_frag(bs, wn * 64 + 16, lane);
    v16bf b2 = lds_b_frag(bs, wn * 64 + 32, lane);
    v16bf b3 = lds_b_frag(bs, wn * 64 + 48, lane);

    acc[0][0] = __builtin_amdgcn_wmma_f32_16x16x32_bf16(false, a0, false, b0, (short)0, acc[0][0], false, false);
    acc[0][1] = __builtin_amdgcn_wmma_f32_16x16x32_bf16(false, a0, false, b1, (short)0, acc[0][1], false, false);
    acc[0][2] = __builtin_amdgcn_wmma_f32_16x16x32_bf16(false, a0, false, b2, (short)0, acc[0][2], false, false);
    acc[0][3] = __builtin_amdgcn_wmma_f32_16x16x32_bf16(false, a0, false, b3, (short)0, acc[0][3], false, false);
    acc[1][0] = __builtin_amdgcn_wmma_f32_16x16x32_bf16(false, a1, false, b0, (short)0, acc[1][0], false, false);
    acc[1][1] = __builtin_amdgcn_wmma_f32_16x16x32_bf16(false, a1, false, b1, (short)0, acc[1][1], false, false);
    acc[1][2] = __builtin_amdgcn_wmma_f32_16x16x32_bf16(false, a1, false, b2, (short)0, acc[1][2], false, false);
    acc[1][3] = __builtin_amdgcn_wmma_f32_16x16x32_bf16(false, a1, false, b3, (short)0, acc[1][3], false, false);
  }

  // Epilogue. C layout (ISA): lane n = lane&15, half = lane>>4, VGPR r -> row half*8+r.
  int row0 = blockIdx.x * 128 + wm * 32;
  int col0 = blockIdx.y * 128 + wn * 64;
  int nlo = lane & 15, half = (lane >> 4) & 1;
#pragma unroll
  for (int mi = 0; mi < 2; ++mi) {
#pragma unroll
    for (int j = 0; j < 4; ++j) {
      int col = col0 + j * 16 + nlo;
      float bv = bias ? bias[col] : 0.f;
#pragma unroll
      for (int r = 0; r < 8; ++r) {
        int row = row0 + mi * 16 + half * 8 + r;
        float v = acc[mi][j][r] + bv;
        if (Cf) Cf[(long)z * cf_mod + (long)row * ldcf + col] = relu_f ? fmaxf(v, 0.f) : v;
        if (Cb) Cb[(long)z * cb_mod + (long)row * ldcb + col] = (__bf16)(relu_b ? fmaxf(v, 0.f) : v);
      }
    }
  }
}

// ---------------- small helper kernels --------------------------------------
__global__ void k_cvt_bf16(const float* __restrict__ in, __bf16* __restrict__ out, long n) {
  long i = (long)blockIdx.x * blockDim.x + threadIdx.x;
  if (i < n) out[i] = (__bf16)in[i];
}

// W (batched) [mat][K][N] fp32 -> Wt [mat][N][K] bf16
__global__ void k_transpose_cvt(const float* __restrict__ W, __bf16* __restrict__ Wt,
                                int K, int N, long total) {
  long i = (long)blockIdx.x * blockDim.x + threadIdx.x;
  if (i >= total) return;
  long mat = i / (long)(K * N);
  long r   = i % (long)(K * N);
  int k = (int)(r / N), n = (int)(r % N);
  Wt[mat * (long)K * N + (long)n * K + k] = (__bf16)W[i];
}

// task_pre[b,e] = relu( (task_idx[b,:] @ z_w[:,e] + z_b[e]) * obs_e[b,e] )  -> bf16
__global__ void k_task_pre(const float* __restrict__ task_idx, const float* __restrict__ z_w,
                           const float* __restrict__ z_b, const float* __restrict__ obs_e,
                           __bf16* __restrict__ out) {
  __shared__ float ti[64];
  int b = blockIdx.x, t = threadIdx.x;
  if (t < NTASK) ti[t] = task_idx[(long)b * NTASK + t];
  __syncthreads();
  for (int e = t; e < EE; e += 256) {
    float a = z_b[e];
#pragma unroll 5
    for (int k = 0; k < NTASK; ++k) a += ti[k] * z_w[k * EE + e];
    float v = a * obs_e[(long)b * EE + e];
    out[(long)b * EE + e] = (__bf16)fmaxf(v, 0.f);
  }
}

// routing for one layer; one 64-thread workgroup per batch row.
__global__ void k_routing(const float* __restrict__ task_e, const float* __restrict__ wcat,
                          int prev_len, const float* __restrict__ pe_w,
                          const float* __restrict__ pe_b,
                          const float* __restrict__ po_w, const float* __restrict__ po_b,
                          int nout, float* __restrict__ probs_out, int out_stride) {
  __shared__ float pv[192];
  __shared__ float te[RE];
  __shared__ float lg[64];
  int b = blockIdx.x, t = threadIdx.x;
  for (int j = t; j < prev_len; j += 64) pv[j] = wcat[(long)b * 192 + j];
  __syncthreads();
  for (int c = t; c < RE; c += 64) {
    float v;
    if (prev_len == 0) {
      v = task_e[(long)b * RE + c];
    } else {
      float a = pe_b[c];
      for (int j = 0; j < prev_len; ++j) a += pv[j] * pe_w[j * RE + c];
      v = task_e[(long)b * RE + c] * a;
    }
    te[c] = fmaxf(v, 0.f);
  }
  __syncthreads();
  if (t < nout) {
    float a = po_b[t];
#pragma unroll 8
    for (int c = 0; c < RE; ++c) a += te[c] * po_w[c * nout + t];
    lg[t] = a;
  }
  __syncthreads();
  if (t < nout) {
    int g = t & ~7;
    float mx = lg[g];
#pragma unroll
    for (int j = 1; j < 8; ++j) mx = fmaxf(mx, lg[g + j]);
    float s = 0.f;
#pragma unroll
    for (int j = 0; j < 8; ++j) s += __expf(lg[g + j] - mx);
    probs_out[(long)b * out_stride + t] = __expf(lg[t] - mx) / s;
  }
}

// module_ins[b,m,o] = relu( sum_n probs[b,m,n] * mod[b,n,o] )
__global__ void k_mix(const __bf16* __restrict__ mod, const float* __restrict__ wcat,
                      int layer, __bf16* __restrict__ mins) {
  __shared__ float p[64];
  int b = blockIdx.x, t = threadIdx.x;
  if (t < 64) p[t] = wcat[(long)b * 192 + layer * 64 + t];
  __syncthreads();
  const __bf16* mb = mod + (long)b * (MM * DD);
  for (int o = t; o < DD; o += 256) {
    float mv[MM];
#pragma unroll
    for (int n = 0; n < MM; ++n) mv[n] = (float)mb[n * DD + o];
#pragma unroll
    for (int m = 0; m < MM; ++m) {
      float a = 0.f;
#pragma unroll
      for (int n = 0; n < MM; ++n) a += p[m * MM + n] * mv[n];
      mins[(long)b * (MM * DD) + m * DD + o] = (__bf16)fmaxf(a, 0.f);
    }
  }
}

// out_e[b,o] = relu( sum_m mod[b,m,o] * probs[b,m] ) -> bf16
__global__ void k_mix_last(const __bf16* __restrict__ mod, const float* __restrict__ plast,
                           __bf16* __restrict__ oute) {
  __shared__ float p[MM];
  int b = blockIdx.x, t = threadIdx.x;
  if (t < MM) p[t] = plast[(long)b * MM + t];
  __syncthreads();
  const __bf16* mb = mod + (long)b * (MM * DD);
  for (int o = t; o < DD; o += 256) {
    float a = 0.f;
#pragma unroll
    for (int m = 0; m < MM; ++m) a += p[m] * (float)mb[m * DD + o];
    oute[(long)b * DD + o] = (__bf16)fmaxf(a, 0.f);
  }
}

// ---------------- host-side orchestration -----------------------------------
extern "C" void kernel_launch(void* const* d_in, const int* in_sizes, int n_in,
                              void* d_out, int out_size, void* d_ws, size_t ws_size,
                              hipStream_t stream) {
  const float* x        = (const float*)d_in[0];
  const float* task_idx = (const float*)d_in[1];
  const float* f_w1 = (const float*)d_in[2];  const float* f_b1 = (const float*)d_in[3];
  const float* f_w2 = (const float*)d_in[4];  const float* f_b2 = (const float*)d_in[5];
  const float* z_w  = (const float*)d_in[6];  const float* z_b  = (const float*)d_in[7];
  const float* te_w1 = (const float*)d_in[8];  const float* te_b1 = (const float*)d_in[9];
  const float* te_w2 = (const float*)d_in[10]; const float* te_b2 = (const float*)d_in[11];
  const float* pe_w[3] = {(const float*)d_in[12], (const float*)d_in[13], (const float*)d_in[14]};
  const float* pe_b[3] = {(const float*)d_in[15], (const float*)d_in[16], (const float*)d_in[17]};
  const float* po_w[4] = {(const float*)d_in[18], (const float*)d_in[19],
                          (const float*)d_in[20], (const float*)d_in[21]};
  const float* po_b[4] = {(const float*)d_in[22], (const float*)d_in[23],
                          (const float*)d_in[24], (const float*)d_in[25]};
  const float* mw = (const float*)d_in[26];
  const float* mb = (const float*)d_in[27];
  const float* head_w = (const float*)d_in[28];
  const float* head_b = (const float*)d_in[29];
  float* out = (float*)d_out;

  // ---- workspace carve ----
  char* wp = (char*)d_ws;
  auto carve = [&](size_t bytes) -> char* {
    char* p = wp; wp += (bytes + 255) & ~(size_t)255; return p;
  };
  __bf16* x_bf    = (__bf16*)carve((size_t)BB * OBS * 2);
  __bf16* h1      = (__bf16*)carve((size_t)BB * 256 * 2);
  float*  obs_e   = (float*) carve((size_t)BB * EE * 4);
  __bf16* obs_rel = (__bf16*)carve((size_t)BB * EE * 2);
  __bf16* tpre    = (__bf16*)carve((size_t)BB * EE * 2);
  __bf16* th1     = (__bf16*)carve((size_t)BB * RE * 2);
  float*  task_e  = (float*) carve((size_t)BB * RE * 4);
  float*  wcat    = (float*) carve((size_t)BB * 192 * 4);
  float*  plast   = (float*) carve((size_t)BB * MM * 4);
  __bf16* modb    = (__bf16*)carve((size_t)BB * MM * DD * 2);
  __bf16* mins    = (__bf16*)carve((size_t)BB * MM * DD * 2);
  __bf16* fw1T  = (__bf16*)carve((size_t)OBS * 256 * 2);
  __bf16* fw2T  = (__bf16*)carve((size_t)256 * EE * 2);
  __bf16* tw1T  = (__bf16*)carve((size_t)EE * RE * 2);
  __bf16* tw2T  = (__bf16*)carve((size_t)RE * RE * 2);
  __bf16* mwT   = (__bf16*)carve((size_t)LL * MM * DD * DD * 2);
  __bf16* hwT   = (__bf16*)carve((size_t)DD * DD * 2);

  auto blk1d = [](long n) { return dim3((unsigned)((n + 255) / 256)); };

  // ---- conversions / transposes ----
  { long n = (long)BB * OBS; k_cvt_bf16<<<blk1d(n), 256, 0, stream>>>(x, x_bf, n); }
  { long n = (long)OBS * 256; k_transpose_cvt<<<blk1d(n), 256, 0, stream>>>(f_w1, fw1T, OBS, 256, n); }
  { long n = (long)256 * EE;  k_transpose_cvt<<<blk1d(n), 256, 0, stream>>>(f_w2, fw2T, 256, EE, n); }
  { long n = (long)EE * RE;   k_transpose_cvt<<<blk1d(n), 256, 0, stream>>>(te_w1, tw1T, EE, RE, n); }
  { long n = (long)RE * RE;   k_transpose_cvt<<<blk1d(n), 256, 0, stream>>>(te_w2, tw2T, RE, RE, n); }
  { long n = (long)DD * DD;   k_transpose_cvt<<<blk1d(n), 256, 0, stream>>>(head_w, hwT, DD, DD, n); }
  { long n = (long)LL * MM * DD * DD;
    k_transpose_cvt<<<blk1d(n), 256, 0, stream>>>(mw, mwT, DD, DD, n); }

  // ---- obs embedding: h1 = relu(x@f_w1+b1); obs_e = h1@f_w2+b2 ----
  gemm_bf16_wmma<<<dim3(BB / 128, 256 / 128, 1), 256, 0, stream>>>(
      x_bf, OBS, 0, fw1T, OBS, 0, f_b1, 0,
      nullptr, 0, 0, 0, h1, 256, 0, 1);
  gemm_bf16_wmma<<<dim3(BB / 128, EE / 128, 1), 256, 0, stream>>>(
      h1, 256, 0, fw2T, 256, 0, f_b2, 0,
      obs_e, EE, 0, 0, obs_rel, EE, 0, 1);

  // ---- task embedding ----
  k_task_pre<<<BB, 256, 0, stream>>>(task_idx, z_w, z_b, obs_e, tpre);
  gemm_bf16_wmma<<<dim3(BB / 128, RE / 128, 1), 256, 0, stream>>>(
      tpre, EE, 0, tw1T, EE, 0, te_b1, 0,
      nullptr, 0, 0, 0, th1, RE, 0, 1);
  gemm_bf16_wmma<<<dim3(BB / 128, RE / 128, 1), 256, 0, stream>>>(
      th1, RE, 0, tw2T, RE, 0, te_b2, 0,
      task_e, RE, 0, 0, nullptr, 0, 0, 0);

  // ---- layers 0..2: routing -> 8 module GEMMs -> 8x8 mix ----
  for (int i = 0; i < LL - 1; ++i) {
    k_routing<<<BB, 64, 0, stream>>>(
        task_e, wcat, i * 64,
        (i == 0) ? nullptr : pe_w[i - 1], (i == 0) ? nullptr : pe_b[i - 1],
        po_w[i], po_b[i], 64, wcat + i * 64, 192);

    const __bf16* A = (i == 0) ? obs_rel : mins;
    int lda         = (i == 0) ? EE : (MM * DD);
    long amod       = (i == 0) ? 0 : DD;
    gemm_bf16_wmma<<<dim3(BB / 128, DD / 128, MM), 256, 0, stream>>>(
        A, lda, amod,
        mwT + (size_t)i * MM * DD * DD, DD, (long)DD * DD,
        mb + (size_t)i * MM * DD, DD,
        nullptr, 0, 0, 0, modb, MM * DD, DD, 0);

    k_mix<<<BB, 256, 0, stream>>>(modb, wcat, i, mins);
  }

  // ---- last layer ----
  k_routing<<<BB, 64, 0, stream>>>(task_e, wcat, 192, pe_w[2], pe_b[2],
                                   po_w[3], po_b[3], MM, plast, MM);
  gemm_bf16_wmma<<<dim3(BB / 128, DD / 128, MM), 256, 0, stream>>>(
      mins, MM * DD, DD,
      mwT + (size_t)(LL - 1) * MM * DD * DD, DD, (long)DD * DD,
      mb + (size_t)(LL - 1) * MM * DD, DD,
      nullptr, 0, 0, 0, modb, MM * DD, DD, 0);
  k_mix_last<<<BB, 256, 0, stream>>>(modb, plast, mins /* reuse as out_e bf16 */);

  // ---- head: out = out_e @ head_w + head_b (fp32) ----
  gemm_bf16_wmma<<<dim3(BB / 128, DD / 128, 1), 256, 0, stream>>>(
      mins, DD, 0, hwT, DD, 0, head_b, 0,
      out, DD, 0, 0, nullptr, 0, 0, 0);
}
